// GAT_1228360647038
// MI455X (gfx1250) — compile-verified, matrix operand order checked by reading, not run on hardware
//
#include <hip/hip_runtime.h>
#include <math.h>

typedef __attribute__((ext_vector_type(16))) _Float16 v16h;
typedef __attribute__((ext_vector_type(8)))  _Float16 v8h;
typedef __attribute__((ext_vector_type(8)))  float    v8f;
typedef __attribute__((ext_vector_type(4)))  unsigned v4u;
typedef __attribute__((ext_vector_type(8)))  int      v8i;
typedef __attribute__((ext_vector_type(4)))  int      v4i;

#define HC     128   // H*C
#define NHEAD  4
#define WPITCH 136   // 128 halfs + 8-half pad per row in LDS (bank decorrelation)

// ---- order-preserving float<->uint mapping for atomicMax-based segment max ----
__device__ __forceinline__ unsigned enc_f32(float f) {
    unsigned u = __float_as_uint(f);
    return (u & 0x80000000u) ? ~u : (u | 0x80000000u);
}
__device__ __forceinline__ float dec_f32(unsigned u) {
    unsigned b = (u & 0x80000000u) ? (u & 0x7FFFFFFFu) : ~u;
    return __uint_as_float(b);
}

// ============================================================================
// Kernel 0: Wt[col*128 + k] = (f16) W[k*128 + col]  (transposed f16 copy of W)
// ============================================================================
__global__ void gat_w_to_f16t(const float* __restrict__ W,
                              _Float16* __restrict__ Wt) {
    const int t   = blockIdx.x * blockDim.x + threadIdx.x;   // 16384 threads
    const int k   = t >> 7;
    const int col = t & 127;
    Wt[col * HC + k] = (_Float16)W[t];
}

// ============================================================================
// Kernel 1: H = X @ W   (N x 128) @ (128 x 128), f16 inputs, f32 accumulate.
//
// W (pre-transposed f16, 32 KB) is DMA'd into LDS once per block by the
// Tensor Data Mover with pad_enable inserting 4 dwords per 64-dword row
// (row pitch 136 halfs).  One wave per 16-row strip; 8 waves -> 128 rows/blk.
//
// WMMA 16x16x32 f16 fragment layout per CDNA5 ISA 7.12.2:
//   A: lane L holds row M=L&15; elems 0-7  -> K = kk + 8*(L>>4) + i
//                               elems 8-15 -> K = kk + 16 + 8*(L>>4) + (i-8)
//   B: lane L holds col N=L&15; same K mapping  (contiguous in LDS -> b128)
//   C/D f32: VGPR r -> M = r + 8*(L>>4), N = L&15.
// ============================================================================
__global__ void gat_gemm_wmma(const float* __restrict__ X,
                              const _Float16* __restrict__ Wt,
                              float* __restrict__ H,
                              int N) {
    __shared__ _Float16 lds_wt[HC * WPITCH];   // 34,816 B

    const int lane = threadIdx.x & 31;
    const int wave = threadIdx.x >> 5;
    const int half = lane >> 4;
    const int l15  = lane & 15;
    const int m0   = blockIdx.x * 128 + wave * 16;

    // ---- wave 0: TDM async copy of Wt (128x128 f16) into padded LDS ----
    if (wave == 0) {
        const unsigned long long ga = (unsigned long long)(uintptr_t)Wt;
        const unsigned lds_base = (unsigned)(uintptr_t)&lds_wt[0];
        v4u g0;
        g0[0] = 1u;                                   // count=1, user descriptor
        g0[1] = lds_base;                             // lds_addr (bytes)
        g0[2] = (unsigned)(ga & 0xFFFFFFFFull);       // global_addr[31:0]
        g0[3] = ((unsigned)(ga >> 32) & 0x01FFFFFFu)  // global_addr[56:32]
              | (2u << 30);                           // type = 2 ("image")
        v8i g1;
        g1[0] = (int)((1u << 16)        // data_size = 1 -> 2-byte elements
                    | (1u << 20)        // pad_enable
                    | (5u << 22)        // pad_interval: 64 dwords (one row)
                    | (3u << 25));      // pad_amount:   4 dwords (8 halfs)
        g1[1] = 128 << 16;              // tensor_dim0 = 128
        g1[2] = 128 << 16;              // tensor_dim1 = 128
        g1[3] = 128 << 16;              // tile_dim0   = 128
        g1[4] = 128;                    // tile_dim1   = 128 (tile_dim2 = 0)
        g1[5] = 128;                    // tensor_dim0_stride = 128
        g1[6] = (int)(16384u << 16);    // tensor_dim1_stride (unused, 2D tile)
        g1[7] = 0;
        v4i gz4 = {0, 0, 0, 0};
        v8i gz8 = {0, 0, 0, 0, 0, 0, 0, 0};
        __builtin_amdgcn_tensor_load_to_lds(g0, g1, gz4, gz4, gz8, 0);
    }

    // ---- preload all 4 A fragments (full K=128) while the DMA runs ----
    int arow = m0 + l15;
    if (arow >= N) arow = N - 1;               // clamp reads; stores guarded
    const float* xr = X + (long long)arow * HC;

    v16h a[4];
#pragma unroll
    for (int ks = 0; ks < 4; ++ks) {
        const int kb1 = ks * 32 + 8 * half;
        const int kb2 = kb1 + 16;
#pragma unroll
        for (int i = 0; i < 8; ++i) a[ks][i]     = (_Float16)xr[kb1 + i];
#pragma unroll
        for (int i = 0; i < 8; ++i) a[ks][8 + i] = (_Float16)xr[kb2 + i];
    }

    __builtin_amdgcn_s_wait_tensorcnt(0);      // TENSORcnt==0 (no-op for waves 1-7)
    __syncthreads();

    const bool full_tile = (m0 + 16 <= N);     // wave-uniform bound check

#pragma unroll
    for (int nt = 0; nt < 8; ++nt) {
        const int col = nt * 16 + l15;
        const _Float16* wl = &lds_wt[col * WPITCH];
        v8f c = {};
#pragma unroll
        for (int ks = 0; ks < 4; ++ks) {
            const int kb1 = ks * 32 + 8 * half;
            const v8h lo = *(const v8h*)(wl + kb1);        // ds_load_b128
            const v8h hi = *(const v8h*)(wl + kb1 + 16);   // ds_load_b128
            const v16h b = __builtin_shufflevector(lo, hi,
                    0, 1, 2, 3, 4, 5, 6, 7, 8, 9, 10, 11, 12, 13, 14, 15);
            c = __builtin_amdgcn_wmma_f32_16x16x32_f16(
                    false, a[ks], false, b, (short)0, c, false, false);
        }
        // D store: branch-free in all blocks except the final partial one.
        float* hp = H + (long long)(m0 + 8 * half) * HC + col;
        if (full_tile) {
#pragma unroll
            for (int r = 0; r < 8; ++r) hp[r * HC] = c[r];
        } else {
#pragma unroll
            for (int r = 0; r < 8; ++r)
                if (m0 + 8 * half + r < N) hp[r * HC] = c[r];
        }
    }
}

// ============================================================================
// Kernel 2: per-node alpha_src/alpha_dst = <h[n,head,:], a_{s,d}[head,:]>,
// plus init of max (encoded -inf -> 0u), den=0, and zeroing the accumulator.
// One wave32 per node; lane owns 4 channels; 8-lane shuffle reduce per head.
// ============================================================================
__global__ void gat_alpha_init(const float* __restrict__ H,
                               const float* __restrict__ avs,
                               const float* __restrict__ avd,
                               float* __restrict__ as,
                               float* __restrict__ ad,
                               unsigned* __restrict__ m,
                               float* __restrict__ den,
                               float* __restrict__ acc,
                               int N) {
    const int lane = threadIdx.x & 31;
    const long long node = ((long long)blockIdx.x * blockDim.x + threadIdx.x) >> 5;
    if (node >= N) return;
    const int c0 = lane * 4;

    const float4 hv = *(const float4*)(H + node * HC + c0);
    const float4 sv = *(const float4*)(avs + c0);
    const float4 dv = *(const float4*)(avd + c0);
    float ps = hv.x * sv.x + hv.y * sv.y + hv.z * sv.z + hv.w * sv.w;
    float pd = hv.x * dv.x + hv.y * dv.y + hv.z * dv.z + hv.w * dv.w;
#pragma unroll
    for (int off = 1; off < 8; off <<= 1) {
        ps += __shfl_xor(ps, off, 32);
        pd += __shfl_xor(pd, off, 32);
    }
    if ((lane & 7) == 0) {
        const int h = lane >> 3;
        as[node * NHEAD + h]  = ps;
        ad[node * NHEAD + h]  = pd;
        m[node * NHEAD + h]   = 0u;      // encodes below any real float
        den[node * NHEAD + h] = 0.0f;
    }
    *(float4*)(acc + node * HC + c0) = make_float4(0.f, 0.f, 0.f, 0.f);
}

// ============================================================================
// Kernel 3: segment max over edges (incl. self-loops) via u32 atomicMax.
// One thread per edge, 4 heads each.  (h/alpha arrays live in the 192 MB L2.)
// ============================================================================
__global__ void gat_edge_max(const int* __restrict__ ei,
                             long long E, int N,
                             const float* __restrict__ as,
                             const float* __restrict__ ad,
                             unsigned* __restrict__ m) {
    const long long e = (long long)blockIdx.x * blockDim.x + threadIdx.x;
    if (e >= E + N) return;
    int src, dst;
    if (e < E) { src = ei[e]; dst = ei[E + e]; }
    else       { src = dst = (int)(e - E); }

    const float4 s = *(const float4*)(as + (long long)src * NHEAD);
    const float4 d = *(const float4*)(ad + (long long)dst * NHEAD);
    float ev[4] = { s.x + d.x, s.y + d.y, s.z + d.z, s.w + d.w };
#pragma unroll
    for (int h = 0; h < 4; ++h) {
        float v = ev[h];
        v = v > 0.f ? v : 0.2f * v;                     // leaky_relu(0.2)
        atomicMax(m + (long long)dst * NHEAD + h, enc_f32(v));
    }
}

// ============================================================================
// Kernel 4: w = exp(e - m[dst]); den[dst,h] += w; acc[dst,:] += w * h[src,:].
// One wave32 per edge; lane owns 4 channels (head = lane>>3).
// ============================================================================
__global__ void gat_edge_accum(const int* __restrict__ ei,
                               long long E, int N,
                               const float* __restrict__ H,
                               const float* __restrict__ as,
                               const float* __restrict__ ad,
                               const unsigned* __restrict__ m,
                               float* __restrict__ den,
                               float* __restrict__ acc) {
    const int lane = threadIdx.x & 31;
    const long long e = ((long long)blockIdx.x * blockDim.x + threadIdx.x) >> 5;
    if (e >= E + N) return;
    int src, dst;
    if (e < E) { src = ei[e]; dst = ei[E + e]; }
    else       { src = dst = (int)(e - E); }

    const int head = lane >> 3;
    float ev = as[(long long)src * NHEAD + head] + ad[(long long)dst * NHEAD + head];
    ev = ev > 0.f ? ev : 0.2f * ev;
    const float mx = dec_f32(m[(long long)dst * NHEAD + head]);
    const float w  = __expf(ev - mx);

    if ((lane & 7) == 0)
        atomicAdd(den + (long long)dst * NHEAD + head, w);

    const int c0 = lane * 4;
    const float4 hv = *(const float4*)(H + (long long)src * HC + c0);
    float* ap = acc + (long long)dst * HC + c0;
    atomicAdd(ap + 0, w * hv.x);
    atomicAdd(ap + 1, w * hv.y);
    atomicAdd(ap + 2, w * hv.z);
    atomicAdd(ap + 3, w * hv.w);
}

// ============================================================================
// Kernel 5: out = relu(acc / den[head] + bias). One wave32 per node.
// ============================================================================
__global__ void gat_normalize(const float* __restrict__ acc,
                              const float* __restrict__ den,
                              const float* __restrict__ bias,
                              float* __restrict__ out,
                              int N) {
    const int lane = threadIdx.x & 31;
    const long long node = ((long long)blockIdx.x * blockDim.x + threadIdx.x) >> 5;
    if (node >= N) return;
    const int c0 = lane * 4;
    const float inv = 1.0f / den[node * NHEAD + (lane >> 3)];
    float4 a = *(const float4*)(acc + node * HC + c0);
    const float4 b = *(const float4*)(bias + c0);
    a.x = fmaxf(a.x * inv + b.x, 0.f);
    a.y = fmaxf(a.y * inv + b.y, 0.f);
    a.z = fmaxf(a.z * inv + b.z, 0.f);
    a.w = fmaxf(a.w * inv + b.w, 0.f);
    *(float4*)(out + node * HC + c0) = a;
}

// ============================================================================
extern "C" void kernel_launch(void* const* d_in, const int* in_sizes, int n_in,
                              void* d_out, int out_size, void* d_ws, size_t ws_size,
                              hipStream_t stream) {
    const float* x    = (const float*)d_in[0];
    const int*   ei   = (const int*)d_in[1];
    const float* W1   = (const float*)d_in[2];
    const float* as1v = (const float*)d_in[3];
    const float* ad1v = (const float*)d_in[4];
    const float* b1   = (const float*)d_in[5];
    const float* W2   = (const float*)d_in[6];
    const float* as2v = (const float*)d_in[7];
    const float* ad2v = (const float*)d_in[8];
    const float* b2   = (const float*)d_in[9];

    const int N       = in_sizes[0] / HC;
    const long long E = in_sizes[1] / 2;
    float* out = (float*)d_out;

    // Workspace carve-up (~109 MB for N=100k)
    char* ws = (char*)d_ws;
    float*     h    = (float*)ws;     ws += (size_t)N * HC * sizeof(float);
    float*     acc1 = (float*)ws;     ws += (size_t)N * HC * sizeof(float);
    float*     as   = (float*)ws;     ws += (size_t)N * NHEAD * sizeof(float);
    float*     ad   = (float*)ws;     ws += (size_t)N * NHEAD * sizeof(float);
    unsigned*  m    = (unsigned*)ws;  ws += (size_t)N * NHEAD * sizeof(unsigned);
    float*     den  = (float*)ws;     ws += (size_t)N * NHEAD * sizeof(float);
    _Float16*  wt   = (_Float16*)ws;  ws += (size_t)HC * HC * sizeof(_Float16);

    const int gemm_blocks      = (N + 127) / 128;
    const int node_wave_blocks = (N + 7) / 8;                 // 8 waves/block
    const long long tot        = E + (long long)N;
    const int edge_thr_blocks  = (int)((tot + 255) / 256);
    const int edge_wave_blocks = (int)((tot + 7) / 8);

    // ---------------- layer 1 ----------------
    gat_w_to_f16t  <<<64,               256, 0, stream>>>(W1, wt);
    gat_gemm_wmma  <<<gemm_blocks,      256, 0, stream>>>(x, wt, h, N);
    gat_alpha_init <<<node_wave_blocks, 256, 0, stream>>>(h, as1v, ad1v, as, ad, m, den, acc1, N);
    gat_edge_max   <<<edge_thr_blocks,  256, 0, stream>>>(ei, E, N, as, ad, m);
    gat_edge_accum <<<edge_wave_blocks, 256, 0, stream>>>(ei, E, N, h, as, ad, m, den, acc1);
    gat_normalize  <<<node_wave_blocks, 256, 0, stream>>>(acc1, den, b1, acc1, N);
    // acc1 now holds relu(layer1) == layer-2 input

    // ---------------- layer 2 ----------------
    gat_w_to_f16t  <<<64,               256, 0, stream>>>(W2, wt);
    gat_gemm_wmma  <<<gemm_blocks,      256, 0, stream>>>(acc1, wt, h, N);
    gat_alpha_init <<<node_wave_blocks, 256, 0, stream>>>(h, as2v, ad2v, as, ad, m, den, out, N);
    gat_edge_max   <<<edge_thr_blocks,  256, 0, stream>>>(ei, E, N, as, ad, m);
    gat_edge_accum <<<edge_wave_blocks, 256, 0, stream>>>(ei, E, N, h, as, ad, m, den, out);
    gat_normalize  <<<node_wave_blocks, 256, 0, stream>>>(out, den, b2, out, N);
}
